// RobertaAdaptedSelfAttention_39264591020599
// MI455X (gfx1250) — matrix operand order repaired and checked by reading, not compiled, not run on hardware
//
#include <hip/hip_runtime.h>
#include <hip/hip_bf16.h>

typedef __attribute__((ext_vector_type(16))) _Float16 v16h;
typedef __attribute__((ext_vector_type(8)))  _Float16 v8h;
typedef __attribute__((ext_vector_type(4)))  _Float16 v4h;
typedef __attribute__((ext_vector_type(8)))  float    v8f;
typedef __attribute__((ext_vector_type(4)))  float    v4f;
typedef __attribute__((ext_vector_type(4)))  int      v4i;

#define B_   4
#define S_   2048
#define HID_ 1024
#define H_   16
#define DH_  64
#define ADP_ 64
#define M_   (B_*S_)

// -------- CDNA5 async global->LDS path (guarded; sync fallback) --------
#if defined(__gfx1250__) && __has_builtin(__builtin_amdgcn_global_load_async_to_lds_b128)
#define HAS_ASYNC_LDS 1
#else
#define HAS_ASYNC_LDS 0
#endif

#define GLOBAL_AS __attribute__((address_space(1)))
#define LDS_AS    __attribute__((address_space(3)))

typedef GLOBAL_AS v4i* gvec_p;
typedef LDS_AS    v4i* lvec_p;

__device__ __forceinline__ void async_b128(const _Float16* g, _Float16* l) {
#if HAS_ASYNC_LDS
  __builtin_amdgcn_global_load_async_to_lds_b128(
      (gvec_p)(uintptr_t)g,
      (lvec_p)(unsigned)(uintptr_t)l, 0, 0);
#else
  *(v8h*)l = *(const v8h*)g;
#endif
}

__device__ __forceinline__ void wait_async0() {
#if HAS_ASYNC_LDS
#if __has_builtin(__builtin_amdgcn_s_wait_asynccnt)
  __builtin_amdgcn_s_wait_asynccnt(0);
#else
  asm volatile("s_wait_asynccnt 0x0" ::: "memory");
#endif
#endif
}

// ---------- WMMA helpers (gfx1250 wave32 fragment layouts) ----------

__device__ __forceinline__ v8f wmma16(v16h a, v16h b, v8f c) {
  return __builtin_amdgcn_wmma_f32_16x16x32_f16(false, a, false, b, (short)0, c,
                                                false, false);
}

__device__ __forceinline__ v16h cat8(v8h lo, v8h hi) {
  return __builtin_shufflevector(lo, hi, 0,1,2,3,4,5,6,7,8,9,10,11,12,13,14,15);
}

// A fragment 16x32 (MxK) from row-major f16 LDS tile.
// lane: m = lane&15 (row), g = lane>>4.  j=0..7 -> k = kbase+8g+j,
// j=8..15 -> k = kbase+16+8g+(j-8).
__device__ __forceinline__ v16h load_a_frag(const _Float16* p, int stride,
                                            int row, int kbase, int g) {
  const _Float16* base = p + row * stride + kbase + 8 * g;
  return cat8(*(const v8h*)base, *(const v8h*)(base + 16));
}

// B fragment 32x16 (KxN) from K-contiguous (transposed) f16 LDS tile.
// lane: n = lane&15 (col), g = lane>>4.  j=0..7 -> k = kbase+8g+j,
// j=8..15 -> k = kbase+16+8g+(j-8).
__device__ __forceinline__ v16h load_b_frag(const _Float16* p, int stride,
                                            int col, int kbase, int g) {
  const _Float16* base = p + col * stride + kbase + 8 * g;
  return cat8(*(const v8h*)base, *(const v8h*)(base + 16));
}

__device__ __forceinline__ float sigm(float x) { return 1.0f / (1.0f + __expf(-x)); }

// =====================================================================
// Kernel 1: QKV projection GEMM  (M=8192, N=1024, K=1024), z = q/k/v
// Block 256 thr (8 waves). Block tile 128x128, BK=32.
// Register-prefetch double buffering: next global tile is in flight
// underneath the 8 WMMAs of the current tile.
// =====================================================================
__global__ __launch_bounds__(256) void qkv_gemm_kernel(
    const float* __restrict__ X,
    const float* __restrict__ Wq, const float* __restrict__ bq,
    const float* __restrict__ Wk, const float* __restrict__ bk,
    const float* __restrict__ Wv, const float* __restrict__ bv,
    _Float16* __restrict__ Qo, _Float16* __restrict__ Ko,
    _Float16* __restrict__ Vo) {
  constexpr int K = HID_, N = HID_;
  __shared__ _Float16 As[128 * 40];  // x tile, row-major [128][32] pad->40
  __shared__ _Float16 Wt[128 * 40];  // W tile transposed [n:128][k:32] pad->40

  const float* Wm;
  const float* bias;
  _Float16* Out;
  if (blockIdx.z == 0)      { Wm = Wq; bias = bq; Out = Qo; }
  else if (blockIdx.z == 1) { Wm = Wk; bias = bk; Out = Ko; }
  else                      { Wm = Wv; bias = bv; Out = Vo; }

  const int m0 = blockIdx.x * 128;
  const int n0 = blockIdx.y * 128;
  const int tid = threadIdx.x;
  const int lane = tid & 31, wave = tid >> 5;
  const int g = lane >> 4, ln = lane & 15;
  const int wm = (wave & 1) * 64;
  const int wn = (wave >> 1) * 32;

  // per-thread staging slices
  const int a_c4 = tid & 7, a_r = tid >> 3;    // A: 4 float4 along k, rows +32
  const int w_c4 = tid & 31, w_kr = tid >> 5;  // W: 4 float4 along n, k +8

  v4f aReg[4], wReg[4];
  auto loadTiles = [&](int kk) {
#pragma unroll
    for (int it = 0; it < 4; ++it)
      aReg[it] = *(const v4f*)(X + (size_t)(m0 + a_r + it * 32) * K + kk + a_c4 * 4);
#pragma unroll
    for (int it = 0; it < 4; ++it)
      wReg[it] = *(const v4f*)(Wm + (size_t)(kk + w_kr + it * 8) * N + n0 + w_c4 * 4);
  };
  auto storeTiles = [&]() {
#pragma unroll
    for (int it = 0; it < 4; ++it) {
      v4h h4;
      h4.x = (_Float16)aReg[it].x; h4.y = (_Float16)aReg[it].y;
      h4.z = (_Float16)aReg[it].z; h4.w = (_Float16)aReg[it].w;
      *(v4h*)(As + (a_r + it * 32) * 40 + a_c4 * 4) = h4;
    }
#pragma unroll
    for (int it = 0; it < 4; ++it) {
      const int krow = w_kr + it * 8;
      Wt[(w_c4 * 4 + 0) * 40 + krow] = (_Float16)wReg[it].x;
      Wt[(w_c4 * 4 + 1) * 40 + krow] = (_Float16)wReg[it].y;
      Wt[(w_c4 * 4 + 2) * 40 + krow] = (_Float16)wReg[it].z;
      Wt[(w_c4 * 4 + 3) * 40 + krow] = (_Float16)wReg[it].w;
    }
  };

  v8f acc[4][2];
#pragma unroll
  for (int mi = 0; mi < 4; ++mi)
#pragma unroll
    for (int ni = 0; ni < 2; ++ni) acc[mi][ni] = (v8f)0.0f;

  loadTiles(0);
  for (int kk = 0; kk < K; kk += 32) {
    storeTiles();
    __syncthreads();
    if (kk + 32 < K) loadTiles(kk + 32);  // prefetch next tile (in flight)

    v16h bfr[2];
#pragma unroll
    for (int ni = 0; ni < 2; ++ni)
      bfr[ni] = load_b_frag(Wt, 40, wn + ni * 16 + ln, 0, g);
    v16h afr[4];
#pragma unroll
    for (int mi = 0; mi < 4; ++mi)
      afr[mi] = load_a_frag(As, 40, wm + mi * 16 + ln, 0, g);
#pragma unroll
    for (int mi = 0; mi < 4; ++mi)
#pragma unroll
      for (int ni = 0; ni < 2; ++ni)
        acc[mi][ni] = wmma16(afr[mi], bfr[ni], acc[mi][ni]);
    __syncthreads();
  }

#pragma unroll
  for (int ni = 0; ni < 2; ++ni) {
    const int col = n0 + wn + ni * 16 + ln;
    const float bv_ = bias[col];
#pragma unroll
    for (int mi = 0; mi < 4; ++mi) {
      const int rowb = m0 + wm + mi * 16 + 8 * g;
#pragma unroll
      for (int v = 0; v < 8; ++v)
        Out[(size_t)(rowb + v) * N + col] = (_Float16)(acc[mi][ni][v] + bv_);
    }
  }
}

// =====================================================================
// Kernel 2: fused adapter  relu(x@fc1+b1)*g1 -> LDS -> relu(h@fc2+b2)*g2
// =====================================================================
__global__ __launch_bounds__(256) void adapter_fused_kernel(
    const float* __restrict__ X,
    const float* __restrict__ fc1w, const float* __restrict__ fc1b,
    const float* __restrict__ fc2w, const float* __restrict__ fc2b,
    const float* __restrict__ efc1, const float* __restrict__ efc2,
    const int* __restrict__ tptr, float* __restrict__ AdOut) {
  constexpr int K = HID_, N = HID_;
  __shared__ _Float16 As[128 * 40];
  __shared__ _Float16 Wt1[64 * 40];
  __shared__ _Float16 Hs[128 * 72];
  __shared__ _Float16 Wt2[128 * 72];

  const int t = *tptr;
  const int m0 = blockIdx.x * 128;
  const int tid = threadIdx.x;
  const int lane = tid & 31, wave = tid >> 5;
  const int g = lane >> 4, ln = lane & 15;
  const int wm = (wave & 1) * 64;

  // ---- phase 1: h = relu(x @ fc1 + b1) * sigmoid(100*efc1[t]) ----
  const int wn16 = (wave >> 1) * 16;
  v8f acc1[4];
#pragma unroll
  for (int mi = 0; mi < 4; ++mi) acc1[mi] = (v8f)0.0f;

  for (int kk = 0; kk < K; kk += 32) {
    {
      const int c4 = tid & 7;
      const int r = tid >> 3;
#pragma unroll
      for (int it = 0; it < 4; ++it) {
        const int row = r + it * 32;
        v4f a = *(const v4f*)(X + (size_t)(m0 + row) * K + kk + c4 * 4);
        v4h h4;
        h4.x = (_Float16)a.x; h4.y = (_Float16)a.y;
        h4.z = (_Float16)a.z; h4.w = (_Float16)a.w;
        *(v4h*)(As + row * 40 + c4 * 4) = h4;
      }
    }
    {
      const int c4 = tid & 15;
      const int kr = tid >> 4;
#pragma unroll
      for (int it = 0; it < 2; ++it) {
        const int krow = kr + it * 16;
        v4f w4 = *(const v4f*)(fc1w + (size_t)(kk + krow) * ADP_ + c4 * 4);
        Wt1[(c4 * 4 + 0) * 40 + krow] = (_Float16)w4.x;
        Wt1[(c4 * 4 + 1) * 40 + krow] = (_Float16)w4.y;
        Wt1[(c4 * 4 + 2) * 40 + krow] = (_Float16)w4.z;
        Wt1[(c4 * 4 + 3) * 40 + krow] = (_Float16)w4.w;
      }
    }
    __syncthreads();
    v16h bfr = load_b_frag(Wt1, 40, wn16 + ln, 0, g);
#pragma unroll
    for (int mi = 0; mi < 4; ++mi) {
      v16h afr = load_a_frag(As, 40, wm + mi * 16 + ln, 0, g);
      acc1[mi] = wmma16(afr, bfr, acc1[mi]);
    }
    __syncthreads();
  }
  {
    const int col1 = wn16 + ln;
    const float gate1 = sigm(100.0f * efc1[t * ADP_ + col1]);
    const float b1 = fc1b[col1];
#pragma unroll
    for (int mi = 0; mi < 4; ++mi) {
      const int rowb = wm + mi * 16 + 8 * g;
#pragma unroll
      for (int v = 0; v < 8; ++v) {
        float val = acc1[mi][v] + b1;
        val = fmaxf(val, 0.0f) * gate1;
        Hs[(rowb + v) * 72 + col1] = (_Float16)val;
      }
    }
  }
  __syncthreads();

  // ---- phase 2: out = relu(h @ fc2 + b2) * sigmoid(100*efc2[t]) ----
  const int wn = (wave >> 1) * 32;
  for (int nc = 0; nc < 8; ++nc) {
    const int n0 = nc * 128;
    {
      const int c4 = tid & 31;
      const int kr = tid >> 5;
#pragma unroll
      for (int it = 0; it < 8; ++it) {
        const int krow = kr + it * 8;
        v4f w4 = *(const v4f*)(fc2w + (size_t)krow * N + n0 + c4 * 4);
        Wt2[(c4 * 4 + 0) * 72 + krow] = (_Float16)w4.x;
        Wt2[(c4 * 4 + 1) * 72 + krow] = (_Float16)w4.y;
        Wt2[(c4 * 4 + 2) * 72 + krow] = (_Float16)w4.z;
        Wt2[(c4 * 4 + 3) * 72 + krow] = (_Float16)w4.w;
      }
    }
    __syncthreads();

    v8f acc[4][2];
#pragma unroll
    for (int mi = 0; mi < 4; ++mi)
#pragma unroll
      for (int ni = 0; ni < 2; ++ni) acc[mi][ni] = (v8f)0.0f;

#pragma unroll
    for (int ks = 0; ks < 64; ks += 32) {
      v16h bfr[2];
#pragma unroll
      for (int ni = 0; ni < 2; ++ni)
        bfr[ni] = load_b_frag(Wt2, 72, wn + ni * 16 + ln, ks, g);
#pragma unroll
      for (int mi = 0; mi < 4; ++mi) {
        v16h afr = load_a_frag(Hs, 72, wm + mi * 16 + ln, ks, g);
#pragma unroll
        for (int ni = 0; ni < 2; ++ni)
          acc[mi][ni] = wmma16(afr, bfr[ni], acc[mi][ni]);
      }
    }
#pragma unroll
    for (int ni = 0; ni < 2; ++ni) {
      const int col = n0 + wn + ni * 16 + ln;
      const float gate2 = sigm(100.0f * efc2[t * N + col]);
      const float b2 = fc2b[col];
#pragma unroll
      for (int mi = 0; mi < 4; ++mi) {
        const int rowb = m0 + wm + mi * 16 + 8 * g;
#pragma unroll
        for (int v = 0; v < 8; ++v) {
          float val = acc[mi][ni][v] + b2;
          val = fmaxf(val, 0.0f) * gate2;
          AdOut[(size_t)(rowb + v) * N + col] = val;
        }
      }
    }
    __syncthreads();
  }
}

// =====================================================================
// Kernel 3: flash attention + adapter add.
// Grid (S/64, B*H), block 128 thr (4 waves); each wave = 16 query rows.
// KV chunks of 32 double-buffered in LDS; K staged with the CDNA5
// async global->LDS path (ASYNCcnt), V through registers (transpose).
// =====================================================================
__global__ __launch_bounds__(128) void flash_attn_kernel(
    const _Float16* __restrict__ Q, const _Float16* __restrict__ Kb,
    const _Float16* __restrict__ Vb, const float* __restrict__ AdOut,
    float* __restrict__ Out) {
  __shared__ _Float16 Ks[2][32 * 72];    // K chunk row-major [kv:32][dh:64]
  __shared__ _Float16 Vt[2][64 * 40];    // V chunk transposed [dh:64][kv:32]
  __shared__ _Float16 Ps[4][16 * 40];    // per-wave P scratch [16][32]

  const int bh = blockIdx.y;
  const int b = bh >> 4, h = bh & 15;
  const int q0 = blockIdx.x * 64;
  const int tid = threadIdx.x;
  const int lane = tid & 31, wave = tid >> 5;
  const int g = lane >> 4, ln = lane & 15;

  // Q fragments for this wave's 16 rows (scaled by 1/sqrt(DH))
  const _Float16* qbase = Q + ((size_t)(b * S_ + q0 + wave * 16)) * HID_ + h * DH_;
  v16h qa[2];
#pragma unroll
  for (int c = 0; c < 2; ++c) {
    const _Float16* p = qbase + (size_t)ln * HID_ + c * 32 + 8 * g;
    v8h lo = *(const v8h*)p;
    v8h hi = *(const v8h*)(p + 16);
    lo = lo * (_Float16)0.125f;
    hi = hi * (_Float16)0.125f;
    qa[c] = cat8(lo, hi);
  }

  // staging slices: r = kv row, d0 = dh segment
  const int r = tid >> 2, d0 = (tid & 3) * 16;
  v8h vr0, vr1;

  auto stageK = [&](int kv0, int buf) {
    const _Float16* ksrc = Kb + ((size_t)(b * S_ + kv0 + r)) * HID_ + h * DH_ + d0;
    _Float16* kdst = &Ks[buf][r * 72 + d0];
    async_b128(ksrc, kdst);
    async_b128(ksrc + 8, kdst + 8);
  };
  auto loadV = [&](int kv0) {
    const _Float16* vsrc = Vb + ((size_t)(b * S_ + kv0 + r)) * HID_ + h * DH_ + d0;
    vr0 = *(const v8h*)vsrc;
    vr1 = *(const v8h*)(vsrc + 8);
  };
  auto storeV = [&](int buf) {
#pragma unroll
    for (int i = 0; i < 8; ++i) {
      Vt[buf][(d0 + i) * 40 + r] = vr0[i];
      Vt[buf][(d0 + 8 + i) * 40 + r] = vr1[i];
    }
  };

  v8f ctx[4];
#pragma unroll
  for (int d = 0; d < 4; ++d) ctx[d] = (v8f)0.0f;
  float mrun[8], lrun[8];
#pragma unroll
  for (int v = 0; v < 8; ++v) { mrun[v] = -1e30f; lrun[v] = 0.0f; }

  stageK(0, 0);
  loadV(0);
  int it = 0;
  for (int kv0 = 0; kv0 < S_; kv0 += 32, ++it) {
    const int cur = it & 1, nxt = cur ^ 1;
    wait_async0();        // complete this wave's async K for chunk `it`
    storeV(cur);
    __syncthreads();      // buf[cur] fully staged & visible
    if (kv0 + 32 < S_) {  // prefetch next chunk underneath the math
      stageK(kv0 + 32, nxt);
      loadV(kv0 + 32);
    }

    const _Float16* KsC = Ks[cur];
    const _Float16* VtC = Vt[cur];

    // scores: two 16x16 tiles (kv 0-15 and 16-31), contraction dh=64
    v8f s0 = (v8f)0.0f, s1 = (v8f)0.0f;
#pragma unroll
    for (int c = 0; c < 2; ++c) {
      v16h kb0 = load_b_frag(KsC, 72, ln, c * 32, g);
      v16h kb1 = load_b_frag(KsC, 72, 16 + ln, c * 32, g);
      s0 = wmma16(qa[c], kb0, s0);
      s1 = wmma16(qa[c], kb1, s1);
    }

    // online softmax (rows live along the 16-lane half)
    float mnew[8], scl[8];
#pragma unroll
    for (int v = 0; v < 8; ++v) {
      float tmax = fmaxf(s0[v], s1[v]);
#pragma unroll
      for (int off = 1; off < 16; off <<= 1)
        tmax = fmaxf(tmax, __shfl_xor(tmax, off, 32));
      mnew[v] = fmaxf(mrun[v], tmax);
      scl[v] = __expf(mrun[v] - mnew[v]);
    }
#pragma unroll
    for (int v = 0; v < 8; ++v) {
      float p0 = __expf(s0[v] - mnew[v]);
      float p1 = __expf(s1[v] - mnew[v]);
      s0[v] = p0; s1[v] = p1;
      float rs = p0 + p1;
#pragma unroll
      for (int off = 1; off < 16; off <<= 1) rs += __shfl_xor(rs, off, 32);
      lrun[v] = lrun[v] * scl[v] + rs;
      mrun[v] = mnew[v];
    }
#pragma unroll
    for (int d = 0; d < 4; ++d)
#pragma unroll
      for (int v = 0; v < 8; ++v) ctx[d][v] *= scl[v];

    // pack P (C layout) -> wave-private LDS -> A fragment
    _Float16* pp = Ps[wave];
#pragma unroll
    for (int v = 0; v < 8; ++v) {
      const int row = v + 8 * g;
      pp[row * 40 + ln] = (_Float16)s0[v];
      pp[row * 40 + 16 + ln] = (_Float16)s1[v];
    }
    v16h pa = cat8(*(const v8h*)(pp + ln * 40 + 8 * g),
                   *(const v8h*)(pp + ln * 40 + 16 + 8 * g));

    // ctx += P @ V   (4 dh tiles of 16)
#pragma unroll
    for (int d = 0; d < 4; ++d) {
      v16h vb = load_b_frag(VtC, 40, d * 16 + ln, 0, g);
      ctx[d] = wmma16(pa, vb, ctx[d]);
    }
    __syncthreads();  // all waves done reading buf[cur]
  }

  // epilogue: /l, add adapter output, store f32
#pragma unroll
  for (int d = 0; d < 4; ++d) {
    const int col = h * DH_ + d * 16 + ln;
#pragma unroll
    for (int v = 0; v < 8; ++v) {
      const int row = q0 + wave * 16 + v + 8 * g;
      const size_t idx = ((size_t)(b * S_ + row)) * HID_ + col;
      Out[idx] = ctx[d][v] / lrun[v] + AdOut[idx];
    }
  }
}

// =====================================================================
extern "C" void kernel_launch(void* const* d_in, const int* in_sizes, int n_in,
                              void* d_out, int out_size, void* d_ws,
                              size_t ws_size, hipStream_t stream) {
  (void)in_sizes; (void)n_in; (void)out_size; (void)ws_size;
  const float* x    = (const float*)d_in[0];
  const float* Wq   = (const float*)d_in[1];
  const float* bq   = (const float*)d_in[2];
  const float* Wk   = (const float*)d_in[3];
  const float* bk   = (const float*)d_in[4];
  const float* Wv   = (const float*)d_in[5];
  const float* bv   = (const float*)d_in[6];
  const float* fc1w = (const float*)d_in[7];
  const float* fc1b = (const float*)d_in[8];
  const float* fc2w = (const float*)d_in[9];
  const float* fc2b = (const float*)d_in[10];
  const float* efc1 = (const float*)d_in[11];
  const float* efc2 = (const float*)d_in[12];
  const int*   tptr = (const int*)d_in[13];

  char* ws = (char*)d_ws;
  const size_t qkv_bytes = (size_t)M_ * HID_ * sizeof(_Float16);  // 16 MB each
  _Float16* Qb = (_Float16*)(ws);
  _Float16* Kb = (_Float16*)(ws + qkv_bytes);
  _Float16* Vb = (_Float16*)(ws + 2 * qkv_bytes);
  float*    Ad = (float*)   (ws + 3 * qkv_bytes);                 // 32 MB

  dim3 gq(M_ / 128, HID_ / 128, 3);
  qkv_gemm_kernel<<<gq, 256, 0, stream>>>(x, Wq, bq, Wk, bk, Wv, bv, Qb, Kb, Vb);

  adapter_fused_kernel<<<dim3(M_ / 128), 256, 0, stream>>>(
      x, fc1w, fc1b, fc2w, fc2b, efc1, efc2, tptr, Ad);

  flash_attn_kernel<<<dim3(S_ / 64, B_ * H_), 128, 0, stream>>>(
      Qb, Kb, Vb, Ad, (float*)d_out);
}
